// CrossAttention_69483980915299
// MI455X (gfx1250) — matrix operand order repaired and checked by reading, not compiled, not run on hardware
//
#include <hip/hip_runtime.h>
#include <math.h>

// ---------------------------------------------------------------------------
// Flash cross-attention for MI455X (gfx1250, wave32, WMMA).
//   O = softmax(Q K^T) V,  N=M=8192, D=DV=512, fp32 in/out, bf16 matrix math.
// ---------------------------------------------------------------------------

typedef __bf16 bf16_t;
typedef __attribute__((ext_vector_type(16))) bf16_t v16bf;
typedef __attribute__((ext_vector_type(8)))  bf16_t v8bf;
typedef __attribute__((ext_vector_type(8)))  float  v8f;

#define N_Q      8192
#define N_KV     8192
#define DIM      512
#define DIMV     512
#define BQ       16        // Q rows per workgroup (one WMMA M tile)
#define BK       32        // K/V rows per iteration
#define NWAVES   4
#define NTHREADS 128
#define DCHUNK   128       // D (and DV) slice per wave

static __device__ __forceinline__ float4 ld4(const float* p) {
    return *(const float4*)p;
}

// Pack 16 fp32 -> 16 bf16 fragment lanes (order matches load order).
static __device__ __forceinline__ v16bf cvt16(float4 a0, float4 a1,
                                              float4 b0, float4 b1) {
    v16bf r;
    r[0]  = (bf16_t)a0.x; r[1]  = (bf16_t)a0.y; r[2]  = (bf16_t)a0.z; r[3]  = (bf16_t)a0.w;
    r[4]  = (bf16_t)a1.x; r[5]  = (bf16_t)a1.y; r[6]  = (bf16_t)a1.z; r[7]  = (bf16_t)a1.w;
    r[8]  = (bf16_t)b0.x; r[9]  = (bf16_t)b0.y; r[10] = (bf16_t)b0.z; r[11] = (bf16_t)b0.w;
    r[12] = (bf16_t)b1.x; r[13] = (bf16_t)b1.y; r[14] = (bf16_t)b1.z; r[15] = (bf16_t)b1.w;
    return r;
}

__global__ __launch_bounds__(NTHREADS)
void flash_xattn_bf16_wmma(const float* __restrict__ Qg,
                           const float* __restrict__ Kg,
                           const float* __restrict__ Vg,
                           float* __restrict__ Og) {
    // LDS: V tile transposed (bf16, padded rows so 16B fragment loads are
    // aligned and bank-friendly), score tile (f32), P tile (bf16), row stats.
    __shared__ bf16_t sVt[DIMV][BK + 8];   // 40 KB
    __shared__ float  sS[BQ][BK + 4];      // 2.25 KB
    __shared__ bf16_t sP[BQ][BK + 8];      // 1.25 KB
    __shared__ float  sScale[BQ];
    __shared__ float  sRowM[BQ];
    __shared__ float  sRowL[BQ];

    const int tid   = threadIdx.x;
    const int w     = tid >> 5;     // wave id: owns D columns [w*128, w*128+128)
    const int lane  = tid & 31;
    const int mm    = lane & 15;    // row (A/C) or column (B) within 16-tile
    const int hi16  = lane >> 4;    // lane half selects K-run per ISA layout
    const int qrow0 = blockIdx.x * BQ;

    if (tid < BQ) { sRowM[tid] = -INFINITY; sRowL[tid] = 0.0f; }

    // --- Preload Q A-fragments (fp32 -> bf16) into VGPRs; reused all loop. ---
    // A 16x32 bf16 layout: lane mm = row M, K runs {hi16*8..+7, 16+hi16*8..+7}.
    v16bf qf[4];
    #pragma unroll
    for (int c = 0; c < 4; ++c) {
        const float* p = Qg + (size_t)(qrow0 + mm) * DIM + w * DCHUNK + c * 32 + hi16 * 8;
        qf[c] = cvt16(ld4(p), ld4(p + 4), ld4(p + 16), ld4(p + 20));
    }

    const v8f vzero = {0.f, 0.f, 0.f, 0.f, 0.f, 0.f, 0.f, 0.f};
    v8f acc[8];
    #pragma unroll
    for (int t = 0; t < 8; ++t) acc[t] = vzero;

    for (int j0 = 0; j0 < N_KV; j0 += BK) {
        // Prefetch next K/V block (global_prefetch_b8).
        if (j0 + BK < N_KV) {
            __builtin_prefetch(Kg + (size_t)(j0 + BK + (tid >> 2)) * DIM  + (tid & 3) * 128, 0, 1);
            __builtin_prefetch(Vg + (size_t)(j0 + BK + (tid >> 2)) * DIMV + (tid & 3) * 128, 0, 1);
        }

        // --- Stage V^T tile into LDS as bf16; zero the score tile. ---
        for (int i = tid; i < (BK * DIMV) / 4; i += NTHREADS) {
            const int row = i >> 7;        // DIMV/4 = 128 float4 per V row
            const int c4  = i & 127;
            float4 v = ld4(Vg + (size_t)(j0 + row) * DIMV + c4 * 4);
            const int c = c4 * 4;
            sVt[c + 0][row] = (bf16_t)v.x;
            sVt[c + 1][row] = (bf16_t)v.y;
            sVt[c + 2][row] = (bf16_t)v.z;
            sVt[c + 3][row] = (bf16_t)v.w;
        }
        for (int i = tid; i < BQ * (BK + 4); i += NTHREADS) (&sS[0][0])[i] = 0.0f;
        __syncthreads();

        // --- S = Q K^T: each wave does two 16-col tiles over its D slice. ---
        // B 32x16 bf16 layout: lane mm = column n (a K row), k runs {hi16*8..+7, 16+hi16*8..+7}.
        #pragma unroll
        for (int tt = 0; tt < 2; ++tt) {
            v8f s = vzero;
            const float* kb = Kg + (size_t)(j0 + tt * 16 + mm) * DIM + w * DCHUNK + hi16 * 8;
            #pragma unroll
            for (int c = 0; c < 4; ++c) {
                const float* p = kb + c * 32;
                v16bf bf = cvt16(ld4(p), ld4(p + 4), ld4(p + 16), ld4(p + 20));
                s = __builtin_amdgcn_wmma_f32_16x16x32_bf16(
                        false, qf[c], false, bf, (short)0, s, false, false);
            }
            // C layout: VGPR r -> row r (lanes 0-15) / row r+8 (lanes 16-31).
            const int col = tt * 16 + mm;
            #pragma unroll
            for (int r = 0; r < 8; ++r)
                atomicAdd(&sS[r + 8 * hi16][col], s[r]);   // ds_add_f32 reduce
        }
        __syncthreads();

        // --- Online softmax over the BK new columns (one lane per Q row). ---
        if (tid < BQ) {
            const int mrow = tid;
            float mOld = sRowM[mrow];
            float lOld = sRowL[mrow];
            float tmax = -INFINITY;
            float sv[BK];
            #pragma unroll
            for (int j = 0; j < BK; ++j) {
                sv[j] = sS[mrow][j];
                tmax  = fmaxf(tmax, sv[j]);
            }
            const float mNew = fmaxf(mOld, tmax);
            const float corr = __expf(mOld - mNew);
            float lsum = 0.0f;
            #pragma unroll
            for (int j = 0; j < BK; ++j) {
                const float pj = __expf(sv[j] - mNew);
                lsum += pj;
                sP[mrow][j] = (bf16_t)pj;
            }
            sRowM[mrow]  = mNew;
            sRowL[mrow]  = lOld * corr + lsum;
            sScale[mrow] = corr;
        }
        __syncthreads();

        // --- Rescale accumulators, then O_acc += P @ V (per-wave DV slice). ---
        float sc[8];
        #pragma unroll
        for (int r = 0; r < 8; ++r) sc[r] = sScale[r + 8 * hi16];

        v8bf plo = *(const v8bf*)(&sP[mm][hi16 * 8]);
        v8bf phi = *(const v8bf*)(&sP[mm][16 + hi16 * 8]);
        v16bf pf = __builtin_shufflevector(plo, phi,
                    0, 1, 2, 3, 4, 5, 6, 7, 8, 9, 10, 11, 12, 13, 14, 15);

        #pragma unroll
        for (int t = 0; t < 8; ++t) {
            #pragma unroll
            for (int r = 0; r < 8; ++r) acc[t][r] *= sc[r];
            const int vcol = w * DCHUNK + t * 16 + mm;
            v8bf vlo = *(const v8bf*)(&sVt[vcol][hi16 * 8]);
            v8bf vhi = *(const v8bf*)(&sVt[vcol][16 + hi16 * 8]);
            v16bf vf = __builtin_shufflevector(vlo, vhi,
                        0, 1, 2, 3, 4, 5, 6, 7, 8, 9, 10, 11, 12, 13, 14, 15);
            acc[t] = __builtin_amdgcn_wmma_f32_16x16x32_bf16(
                        false, pf, false, vf, (short)0, acc[t], false, false);
        }
        __syncthreads();   // protect sVt/sS/sP before next iteration restages
    }

    // --- Epilogue: divide by softmax denominator, store fp32 output. ---
    float li[8];
    #pragma unroll
    for (int r = 0; r < 8; ++r) li[r] = 1.0f / sRowL[r + 8 * hi16];
    #pragma unroll
    for (int t = 0; t < 8; ++t) {
        const int col = w * DCHUNK + t * 16 + mm;
        #pragma unroll
        for (int r = 0; r < 8; ++r) {
            const int row = r + 8 * hi16;
            Og[(size_t)(qrow0 + row) * DIMV + col] = acc[t][r] * li[r];
        }
    }
}

extern "C" void kernel_launch(void* const* d_in, const int* in_sizes, int n_in,
                              void* d_out, int out_size, void* d_ws, size_t ws_size,
                              hipStream_t stream) {
    (void)in_sizes; (void)n_in; (void)out_size; (void)d_ws; (void)ws_size;
    const float* Q = (const float*)d_in[0];
    const float* K = (const float*)d_in[1];
    const float* V = (const float*)d_in[2];
    float* O = (float*)d_out;

    dim3 grid(N_Q / BQ);      // 512 workgroups
    dim3 block(NTHREADS);     // 4 wave32s
    hipLaunchKernelGGL(flash_xattn_bf16_wmma, grid, block, 0, stream, Q, K, V, O);
}
